// MambaBlock_77395310673965
// MI455X (gfx1250) — matrix operand order repaired and checked by reading, not compiled, not run on hardware
//
#include <hip/hip_runtime.h>
#include <hip/hip_bf16.h>
#include <math.h>

#define D_MODEL 1024
#define D_STATE 16
#define D_CONV  4
#define D_INNER 2048
#define BATCH   2
#define SEQ     4096
#define MROWS   (BATCH * SEQ)   // 8192

typedef __attribute__((ext_vector_type(16))) __bf16         v16bf;
typedef __attribute__((ext_vector_type(8)))  float          v8f;
typedef __attribute__((ext_vector_type(8)))  unsigned short v8u;
typedef __attribute__((ext_vector_type(16))) unsigned short v16u;
typedef __attribute__((ext_vector_type(4)))  int            v4i;

#if defined(__AMDGCN__) && __has_builtin(__builtin_amdgcn_global_load_async_to_lds_b128)
#define ASYNC_LDS 1
#else
#define ASYNC_LDS 0
#endif

static __device__ __forceinline__ unsigned short f32_to_bf16_rne(float f) {
    union { float f; unsigned u; } v; v.f = f;
    unsigned u = v.u;
    if ((u & 0x7fffffffu) > 0x7f800000u) return (unsigned short)((u >> 16) | 0x40); // quiet NaN
    unsigned r = u + 0x7fffu + ((u >> 16) & 1u);                                    // RNE
    return (unsigned short)(r >> 16);
}

// 16-byte global -> LDS copy; async tensor-feed path on CDNA5, sync fallback.
static __device__ __forceinline__ void cp16(unsigned short* lds, const unsigned short* g) {
#if ASYNC_LDS
    __builtin_amdgcn_global_load_async_to_lds_b128(
        (__attribute__((address_space(1))) v4i*)g,
        (__attribute__((address_space(3))) v4i*)lds, 0, 0);
#else
    *(v8u*)lds = *(const v8u*)g;
#endif
}

template <int N>
static __device__ __forceinline__ void wait_async() {
#if ASYNC_LDS
#if __has_builtin(__builtin_amdgcn_s_wait_asynccnt)
    __builtin_amdgcn_s_wait_asynccnt(N);
#else
    asm volatile("s_wait_asynccnt %0" ::"i"(N) : "memory");
#endif
#endif
}

// ---------------------------------------------------------------------------
// fp32 -> bf16 downconvert (x, in_proj_w, out_proj_w)
// ---------------------------------------------------------------------------
__global__ void cvt_f32_bf16_kernel(const float* __restrict__ src,
                                    unsigned short* __restrict__ dst, int n) {
    int i = blockIdx.x * blockDim.x + threadIdx.x;
    if (i < n) dst[i] = f32_to_bf16_rne(src[i]);
}

// ---------------------------------------------------------------------------
// WMMA bf16 GEMM:  C[M x N] = A[M x K] * W[N x K]^T   (A, W bf16; C f32)
// block = 256 threads = 8 waves (2 x 4), wave tile 32(M) x 64(N), block 64 x 256
// Double-buffered LDS staging via async global->LDS copies.
// grid.x = N/256, grid.y = M/64 ; K % 32 == 0
// ---------------------------------------------------------------------------
template <int K>
__global__ __launch_bounds__(256) void wmma_gemm_bf16_kernel(
        const unsigned short* __restrict__ A,
        const unsigned short* __restrict__ W,
        float* __restrict__ C, int ldc) {
    constexpr int LDS_STRIDE = 40;     // 32 + 8 pad: 16 rows cover all 64 banks once
    __shared__ __align__(16) unsigned short As[2][64 * LDS_STRIDE];    // 2 x 5 KB
    __shared__ __align__(16) unsigned short Bs[2][256 * LDS_STRIDE];   // 2 x 20 KB

    const int tid  = threadIdx.x;
    const int lane = tid & 31;
    const int wid  = tid >> 5;
    const int wm   = wid >> 2;          // 0..1
    const int wn   = wid & 3;           // 0..3
    const int l15  = lane & 15;
    const int lh   = lane >> 4;         // 0 or 1
    const size_t m_blk = (size_t)blockIdx.y * 64;
    const size_t n_blk = (size_t)blockIdx.x * 256;

    // staging assignment: thread -> (row 0..63, 16B chunk 0..3)
    const int srow = tid >> 2;
    const int scol = (tid & 3) * 8;
    const unsigned short* Ag = A + (m_blk + srow) * (size_t)K + scol;
    const unsigned short* Wg = W + (n_blk + srow) * (size_t)K + scol;

    auto stage = [&](int buf, int k0) {
        cp16(&As[buf][srow * LDS_STRIDE + scol], Ag + k0);
#pragma unroll
        for (int c = 0; c < 4; ++c)
            cp16(&Bs[buf][(c * 64 + srow) * LDS_STRIDE + scol],
                 Wg + (size_t)(c * 64) * K + k0);
    };

    v8f acc[2][4];
#pragma unroll
    for (int mi = 0; mi < 2; ++mi)
#pragma unroll
        for (int ni = 0; ni < 4; ++ni)
#pragma unroll
            for (int r = 0; r < 8; ++r) acc[mi][ni][r] = 0.0f;

    stage(0, 0);
    constexpr int KT = K / 32;
    for (int kt = 0; kt < KT; ++kt) {
        const int buf = kt & 1;
        if (kt + 1 < KT) {
            stage(buf ^ 1, (kt + 1) * 32);   // prefetch next tile into other buffer
            wait_async<5>();                 // wait for THIS tile's 5 async ops
        } else {
            wait_async<0>();
        }
        __syncthreads();

        // A fragment (ISA 7.12.2, 16-bit A 16x32):
        // lane<16 : row l15, K {0..7,16..23}; lane>=16: row l15, K {8..15,24..31}
        v16bf afrag[2];
#pragma unroll
        for (int mi = 0; mi < 2; ++mi) {
            const unsigned short* ap =
                &As[buf][(mi * 16 + l15) * LDS_STRIDE + lh * 8];
            v8u lo = *(const v8u*)ap;
            v8u hi = *(const v8u*)(ap + 16);
            v16u t = __builtin_shufflevector(lo, hi, 0, 1, 2, 3, 4, 5, 6, 7,
                                             8, 9, 10, 11, 12, 13, 14, 15);
            afrag[mi] = __builtin_bit_cast(v16bf, t);
        }
        // B fragment (32x16): lane column l15, K {8*lh..8*lh+7, 16+8*lh..16+8*lh+7}
        v16bf bfrag[4];
#pragma unroll
        for (int ni = 0; ni < 4; ++ni) {
            const unsigned short* bp =
                &Bs[buf][(wn * 64 + ni * 16 + l15) * LDS_STRIDE + lh * 8];
            v8u lo = *(const v8u*)bp;
            v8u hi = *(const v8u*)(bp + 16);
            v16u t = __builtin_shufflevector(lo, hi, 0, 1, 2, 3, 4, 5, 6, 7,
                                             8, 9, 10, 11, 12, 13, 14, 15);
            bfrag[ni] = __builtin_bit_cast(v16bf, t);
        }
#pragma unroll
        for (int mi = 0; mi < 2; ++mi)
#pragma unroll
            for (int ni = 0; ni < 4; ++ni)
                acc[mi][ni] = __builtin_amdgcn_wmma_f32_16x16x32_bf16(
                    false, afrag[mi], false, bfrag[ni],
                    (short)0, acc[mi][ni], false, false);
        __syncthreads();   // protect buf before it is overwritten at kt+2
    }

    // C/D layout: element r -> row 16*mi + 8*lh + r, col = 16*ni + l15
#pragma unroll
    for (int mi = 0; mi < 2; ++mi)
#pragma unroll
        for (int ni = 0; ni < 4; ++ni) {
            size_t n = n_blk + (size_t)wn * 64 + (size_t)ni * 16 + l15;
#pragma unroll
            for (int r = 0; r < 8; ++r) {
                size_t m = m_blk + (size_t)wm * 32 + (size_t)mi * 16 +
                           (size_t)lh * 8 + r;
                C[m * (size_t)ldc + n] = acc[mi][ni][r];
            }
        }
}

// ---------------------------------------------------------------------------
// Depthwise causal conv (width 4) over sequence + SiLU.
// xz is (m, 4096) with columns [0,2048) = xc ; writes xs (m, 2048) in f32.
// ---------------------------------------------------------------------------
__global__ void conv_silu_kernel(const float* __restrict__ xz,
                                 const float* __restrict__ conv_w,
                                 const float* __restrict__ conv_b,
                                 float* __restrict__ xs) {
    int i = blockIdx.x * blockDim.x + threadIdx.x;
    if (i >= BATCH * SEQ * D_INNER) return;
    int d = i % D_INNER;
    int s = (i / D_INNER) % SEQ;
    int b = i / (D_INNER * SEQ);
    float acc = conv_b[d];
#pragma unroll
    for (int j = 0; j < D_CONV; ++j) {
        int t = s - (D_CONV - 1) + j;
        if (t >= 0)
            acc += conv_w[d * D_CONV + j] *
                   xz[((size_t)(b * SEQ + t)) * (2 * D_INNER) + d];
    }
    xs[(size_t)i] = acc / (1.0f + __expf(-acc));   // SiLU
}

// ---------------------------------------------------------------------------
// x_proj: dbc[m, j] = dot(xs[m, :], x_proj_w[j, :])   j in [0, 33)
// one wave per (m, j); strided dot + shfl_xor reduction
// ---------------------------------------------------------------------------
__global__ void xproj_kernel(const float* __restrict__ xs,
                             const float* __restrict__ xw,
                             float* __restrict__ dbc) {
    int gw   = (blockIdx.x * blockDim.x + threadIdx.x) >> 5;
    int lane = threadIdx.x & 31;
    if (gw >= MROWS * 33) return;
    int m = gw / 33, j = gw % 33;
    const float* xr = xs + (size_t)m * D_INNER;
    const float* wr = xw + (size_t)j * D_INNER;
    float acc = 0.f;
    for (int k = lane; k < D_INNER; k += 32) acc += xr[k] * wr[k];
#pragma unroll
    for (int off = 16; off > 0; off >>= 1) acc += __shfl_xor(acc, off, 32);
    if (lane == 0) dbc[(size_t)m * 33 + j] = acc;
}

// ---------------------------------------------------------------------------
// Selective scan + z-gate, fused softplus(delta) and bf16 downconvert.
// lane = (dsub<<4) | n : one wave handles 2 channels x 16 states.
// ---------------------------------------------------------------------------
__global__ void scan_gate_kernel(const float* __restrict__ xs,
                                 const float* __restrict__ dbc,
                                 const float* __restrict__ xz,   // z = cols [2048,4096)
                                 const float* __restrict__ dtw,
                                 const float* __restrict__ dtb,
                                 const float* __restrict__ A_log,
                                 const float* __restrict__ Dp,
                                 unsigned short* __restrict__ yg) {
    int gw   = (blockIdx.x * blockDim.x + threadIdx.x) >> 5;
    int lane = threadIdx.x & 31;
    if (gw >= BATCH * (D_INNER / 2)) return;
    int b    = gw / (D_INNER / 2);
    int pair = gw % (D_INNER / 2);
    int dsub = lane >> 4;
    int n    = lane & 15;
    int d    = pair * 2 + dsub;

    float Ad   = -__expf(A_log[d * D_STATE + n]);
    float Dd   = Dp[d];
    float dtwd = dtw[d];
    float dtbd = dtb[d];
    float h    = 0.f;

    for (int t = 0; t < SEQ; ++t) {
        size_t m = (size_t)b * SEQ + t;
        const float* db = dbc + m * 33;
        float draw = db[0];
        float Bn   = db[1 + n];
        float Cn   = db[17 + n];
        float x    = xs[m * D_INNER + d];
        float arg  = draw * dtwd + dtbd;
        float dl   = (arg > 20.f) ? arg : log1pf(__expf(arg));   // softplus
        h = __expf(dl * Ad) * h + (dl * x) * Bn;
        float yn = h * Cn;
        yn += __shfl_xor(yn, 1, 32);
        yn += __shfl_xor(yn, 2, 32);
        yn += __shfl_xor(yn, 4, 32);
        yn += __shfl_xor(yn, 8, 32);                             // sum over 16 states
        if (n == 0) {
            float zv = xz[m * (2 * D_INNER) + D_INNER + d];
            float y  = yn + Dd * x;
            float g  = zv / (1.f + __expf(-zv));                 // SiLU(z)
            yg[m * D_INNER + d] = f32_to_bf16_rne(y * g);
        }
    }
}

// ---------------------------------------------------------------------------
extern "C" void kernel_launch(void* const* d_in, const int* in_sizes, int n_in,
                              void* d_out, int out_size, void* d_ws, size_t ws_size,
                              hipStream_t stream) {
    const float* x          = (const float*)d_in[0];
    const float* in_proj_w  = (const float*)d_in[1];
    const float* conv_w     = (const float*)d_in[2];
    const float* conv_b     = (const float*)d_in[3];
    const float* x_proj_w   = (const float*)d_in[4];
    const float* dt_proj_w  = (const float*)d_in[5];
    const float* dt_proj_b  = (const float*)d_in[6];
    const float* A_log      = (const float*)d_in[7];
    const float* Dp         = (const float*)d_in[8];
    const float* out_proj_w = (const float*)d_in[9];
    float* out = (float*)d_out;

    // workspace layout (all offsets 256B-aligned)
    const size_t SZ_XBF  = (size_t)MROWS * D_MODEL * 2;           // 16 MiB
    const size_t SZ_WIN  = (size_t)(2 * D_INNER) * D_MODEL * 2;   //  8 MiB
    const size_t SZ_WOUT = (size_t)D_MODEL * D_INNER * 2;         //  4 MiB
    const size_t SZ_XZ   = (size_t)MROWS * (2 * D_INNER) * 4;     // 128 MiB
    const size_t SZ_XS   = (size_t)MROWS * D_INNER * 4;           // 64 MiB
    const size_t SZ_DBC  = (size_t)MROWS * 33 * 4;                // ~1 MiB (256-divisible)

    char* ws = (char*)d_ws;
    size_t off = 0;
    unsigned short* xbf   = (unsigned short*)(ws + off); off += SZ_XBF;
    unsigned short* winbf = (unsigned short*)(ws + off); off += SZ_WIN;
    unsigned short* woutbf= (unsigned short*)(ws + off); off += SZ_WOUT;
    float*          xzbuf = (float*)(ws + off);          off += SZ_XZ;
    float*          xsbuf = (float*)(ws + off);          off += SZ_XS;
    float*          dbcbuf= (float*)(ws + off);          off += SZ_DBC;
    unsigned short* ygbuf = (unsigned short*)(ws + off);

    // 1) fp32 -> bf16 for GEMM operands
    {
        int n = MROWS * D_MODEL;
        cvt_f32_bf16_kernel<<<(n + 255) / 256, 256, 0, stream>>>(x, xbf, n);
        n = 2 * D_INNER * D_MODEL;
        cvt_f32_bf16_kernel<<<(n + 255) / 256, 256, 0, stream>>>(in_proj_w, winbf, n);
        n = D_MODEL * D_INNER;
        cvt_f32_bf16_kernel<<<(n + 255) / 256, 256, 0, stream>>>(out_proj_w, woutbf, n);
    }

    // 2) in_proj GEMM: xz[8192 x 4096] = x[8192 x 1024] * Win[4096 x 1024]^T
    {
        dim3 grid((2 * D_INNER) / 256, MROWS / 64);
        wmma_gemm_bf16_kernel<D_MODEL><<<grid, 256, 0, stream>>>(
            xbf, winbf, xzbuf, 2 * D_INNER);
    }

    // 3) depthwise causal conv + SiLU -> xs
    {
        int n = BATCH * SEQ * D_INNER;
        conv_silu_kernel<<<(n + 255) / 256, 256, 0, stream>>>(xzbuf, conv_w, conv_b, xsbuf);
    }

    // 4) x_proj: dbc[8192 x 33]
    {
        long long waves = (long long)MROWS * 33;
        long long thr   = waves * 32;
        xproj_kernel<<<(unsigned)((thr + 255) / 256), 256, 0, stream>>>(
            xsbuf, x_proj_w, dbcbuf);
    }

    // 5) selective scan + gate -> yg (bf16)
    {
        int waves = BATCH * (D_INNER / 2);   // 2048
        int thr   = waves * 32;
        scan_gate_kernel<<<(thr + 255) / 256, 256, 0, stream>>>(
            xsbuf, dbcbuf, xzbuf, dt_proj_w, dt_proj_b, A_log, Dp, ygbuf);
    }

    // 6) out_proj GEMM: out[8192 x 1024] = yg[8192 x 2048] * Wout[1024 x 2048]^T
    {
        dim3 grid(D_MODEL / 256, MROWS / 64);
        wmma_gemm_bf16_kernel<D_INNER><<<grid, 256, 0, stream>>>(
            ygbuf, woutbf, out, D_MODEL);
    }
}
